// CrossAttentionLayer_2611340116256
// MI455X (gfx1250) — matrix operand order repaired and checked by reading, not compiled, not run on hardware
//
#include <hip/hip_runtime.h>
#include <stdint.h>

#define BB  2
#define SS  2048
#define HH  768
#define NHH 12
#define HDD 64
#define MM  (BB * SS)   // 4096 rows

typedef __attribute__((ext_vector_type(16))) __bf16 v16bf;
typedef __attribute__((ext_vector_type(8)))  float  v8f;
typedef __attribute__((ext_vector_type(4))) unsigned int u32x4;
typedef __attribute__((ext_vector_type(8))) int i32x8;
typedef __attribute__((ext_vector_type(4))) int i32x4;

#if defined(__HIP_DEVICE_COMPILE__) && __has_builtin(__builtin_amdgcn_tensor_load_to_lds)
#define HAVE_TDM 1
#else
#define HAVE_TDM 0
#endif

union Frag {
    v16bf    v;
    uint32_t u[8];
};

__device__ __forceinline__ v8f wmma_bf16(const Frag& a, const Frag& b, v8f c) {
    return __builtin_amdgcn_wmma_f32_16x16x32_bf16(false, a.v, false, b.v,
                                                   (short)0, c, false, false);
}

// 16-bit A-matrix 16x32 K-index pattern per (vgpr, lane-half) — ISA 7.12.2
__device__ __forceinline__ int a_k(int v, int half) {
    return (v < 4) ? (half * 8 + v * 2) : (16 + half * 8 + (v - 4) * 2);
}

__device__ __forceinline__ float rmax16(float v) {
    v = fmaxf(v, __shfl_xor(v, 1, 32));
    v = fmaxf(v, __shfl_xor(v, 2, 32));
    v = fmaxf(v, __shfl_xor(v, 4, 32));
    v = fmaxf(v, __shfl_xor(v, 8, 32));
    return v;
}
__device__ __forceinline__ float rsum16(float v) {
    v += __shfl_xor(v, 1, 32);
    v += __shfl_xor(v, 2, 32);
    v += __shfl_xor(v, 4, 32);
    v += __shfl_xor(v, 8, 32);
    return v;
}

// ---- Tensor Data Mover: 2D tile (bf16 elements) global -> LDS --------------
// D# per CDNA5 ISA: group0 = {count|flags, lds_addr,
// global_addr, type=2}; group1 = {wg_mask=0, data_size=2B, dims/tile/stride};
// groups 2/3 zero (2D tensor). Tracked by TENSORcnt.
__device__ __forceinline__ void tdm_load_2d(uint32_t lds_addr, const void* gptr,
                                            uint32_t td0, uint32_t td1,
                                            uint32_t tile_d0, uint32_t tile_d1,
                                            uint32_t stride0) {
#if HAVE_TDM
    uint64_t ga = (uint64_t)(uintptr_t)gptr;
    u32x4 g0;
    g0[0] = 1u;                                                // count=1 (valid), user mode
    g0[1] = lds_addr;                                          // lds_addr (bytes)
    g0[2] = (uint32_t)ga;                                      // global_addr[31:0]
    g0[3] = ((uint32_t)(ga >> 32) & 0x01ffffffu) | (2u << 30); // addr[56:32] | type=2
    i32x8 g1;
    g1[0] = (int)(1u << 16);                                   // data_size=1 (2 bytes)
    g1[1] = (int)((td0 & 0xffffu) << 16);                      // tensor_dim0[15:0]
    g1[2] = (int)(((td0 >> 16) & 0xffffu) | ((td1 & 0xffffu) << 16));
    g1[3] = (int)(((td1 >> 16) & 0xffffu) | ((tile_d0 & 0xffffu) << 16));
    g1[4] = (int)(tile_d1 & 0xffffu);                          // tile_dim1, tile_dim2=0
    g1[5] = (int)stride0;                                      // tensor_dim0_stride[31:0]
    g1[6] = 0;                                                 // stride0[47:32], stride1[15:0]
    g1[7] = 0;
    i32x4 z4 = {0, 0, 0, 0};
#if defined(__clang_major__) && (__clang_major__ >= 23)
    i32x8 z8 = {0, 0, 0, 0, 0, 0, 0, 0};
    __builtin_amdgcn_tensor_load_to_lds(g0, g1, z4, z4, z8, 0);
#else
    __builtin_amdgcn_tensor_load_to_lds(g0, g1, z4, z4, 0);
#endif
#else
    (void)lds_addr; (void)gptr; (void)td0; (void)td1;
    (void)tile_d0; (void)tile_d1; (void)stride0;
#endif
}

__device__ __forceinline__ uint32_t lds_off(const void* p) {
    // generic pointer to LDS: low 32 bits are the LDS byte address
    return (uint32_t)(uintptr_t)p;
}

// ---------------- LayerNorm + bf16 convert: one block per row ----------------
__global__ void __launch_bounds__(256)
ln_kernel(const float* __restrict__ x, const float* __restrict__ g,
          const float* __restrict__ bta, __bf16* __restrict__ y) {
    const int row = blockIdx.x;
    const int tid = threadIdx.x;
    const float* xr = x + (size_t)row * HH;
    float v0 = xr[tid], v1 = xr[tid + 256], v2 = xr[tid + 512];

    __shared__ float red[256];
    red[tid] = v0 + v1 + v2;
    __syncthreads();
    for (int off = 128; off > 0; off >>= 1) {
        if (tid < off) red[tid] += red[tid + off];
        __syncthreads();
    }
    const float mu = red[0] * (1.0f / HH);
    __syncthreads();
    float d0 = v0 - mu, d1 = v1 - mu, d2 = v2 - mu;
    red[tid] = d0 * d0 + d1 * d1 + d2 * d2;
    __syncthreads();
    for (int off = 128; off > 0; off >>= 1) {
        if (tid < off) red[tid] += red[tid + off];
        __syncthreads();
    }
    const float rstd = rsqrtf(red[0] * (1.0f / HH) + 1e-5f);

    __bf16* yr = y + (size_t)row * HH;
    yr[tid]       = (__bf16)(d0 * rstd * g[tid]       + bta[tid]);
    yr[tid + 256] = (__bf16)(d1 * rstd * g[tid + 256] + bta[tid + 256]);
    yr[tid + 512] = (__bf16)(d2 * rstd * g[tid + 512] + bta[tid + 512]);
}

// ---------------- fp32 -> bf16 elementwise convert ----------------
__global__ void __launch_bounds__(256)
cvt_kernel(const float* __restrict__ src, __bf16* __restrict__ dst, int n) {
    int i = blockIdx.x * 256 + threadIdx.x;
    if (i < n) dst[i] = (__bf16)src[i];
}

// ---------------- WMMA GEMM: [MM,HH] x [HH,HH], mode-specific epilogue -------
// W k-chunk tiles (32x128 bf16 = 8KB) staged into LDS by the TDM, double
// buffered; all 8 waves build B fragments from LDS instead of 8x-redundant
// global loads. All 8 B fragments are preloaded into registers so the
// ds_load_b128 clause issues once and the 8 WMMAs run back-to-back.
// mode 0: Q  -> bf16 [B,NH,S,HD]
// mode 1: K  -> (+0.5*mem0) bf16 transposed [B,NH,HD,S]
// mode 2: V  -> (+0.5*mem1) bf16 [B,NH,S,HD]
// mode 3: O  -> gate/bias/dyn fused, fp32 [B,S,H]
__global__ void __launch_bounds__(256)
gemm_bf16_kernel(const __bf16* __restrict__ A, const __bf16* __restrict__ W,
                 const float* __restrict__ bias, int mode,
                 const float* __restrict__ mem,
                 const float* __restrict__ gate, const float* __restrict__ gbias,
                 const float* __restrict__ dyn,
                 __bf16* __restrict__ obf, float* __restrict__ of32) {
    const int lane = threadIdx.x & 31;
    const int wave = threadIdx.x >> 5;
    const int half = lane >> 4;
    const int l16  = lane & 15;
    const int m0   = blockIdx.x * 128 + wave * 16;   // this wave's 16 rows
    const int n0   = blockIdx.y * 128;               // block's 128 cols

    __shared__ alignas(16) __bf16 wtile[2][32][128];

    v8f zero = {};
    v8f acc[8];
#pragma unroll
    for (int i = 0; i < 8; ++i) acc[i] = zero;

    const __bf16* Arow = A + (size_t)(m0 + l16) * HH;   // per-lane A row (M = l16)

#if HAVE_TDM
    if (wave == 0)
        tdm_load_2d(lds_off(&wtile[0][0][0]), W + n0, HH, HH, 128, 32, HH);
#endif

    for (int it = 0; it < HH / 32; ++it) {
        const int kc  = it * 32;
        const int cur = it & 1;
#if HAVE_TDM
        if (wave == 0) __builtin_amdgcn_s_wait_tensorcnt((short)0);
        __syncthreads();   // publish tile `cur`; order last readers of `cur^1`
        if (wave == 0 && kc + 32 < HH)
            tdm_load_2d(lds_off(&wtile[cur ^ 1][0][0]),
                        W + (size_t)(kc + 32) * HH + n0, HH, HH, 128, 32, HH);
#else
        __syncthreads();
        {
            const __bf16* src = W + (size_t)kc * HH + n0;
            uint32_t* dst = (uint32_t*)&wtile[cur][0][0];
            for (int i = threadIdx.x; i < 32 * 64; i += 256) {
                int r = i >> 6, c = i & 63;
                dst[i] = *reinterpret_cast<const uint32_t*>(src + (size_t)r * HH + c * 2);
            }
        }
        __syncthreads();
#endif
        __builtin_prefetch((const void*)(Arow + kc + 64), 0, 3);
        Frag af;
#pragma unroll
        for (int v = 0; v < 8; ++v)
            af.u[v] = *reinterpret_cast<const uint32_t*>(Arow + kc + a_k(v, half));
        // preload all 8 B fragments from LDS (lane = N col, vgpr packs K pair in a_k order)
        const __bf16* Brow = &wtile[cur][0][l16];
        Frag bf[8];
#pragma unroll
        for (int nb = 0; nb < 8; ++nb)
#pragma unroll
            for (int v = 0; v < 8; ++v)
                for (int j = 0; j < 2; ++j)
                    bf[nb].v[2 * v + j] = Brow[(a_k(v, half) + j) * 128 + nb * 16];
#pragma unroll
        for (int nb = 0; nb < 8; ++nb)
            acc[nb] = wmma_bf16(af, bf[nb], acc[nb]);
    }

#pragma unroll
    for (int nb = 0; nb < 8; ++nb) {
#pragma unroll
        for (int r = 0; r < 8; ++r) {
            const int row = m0 + r + 8 * half;          // C/D: vgpr r -> M row
            const int col = n0 + nb * 16 + l16;         // lane -> N col
            float val = acc[nb][r] + bias[col];
            const int b = row >> 11, s = row & (SS - 1);
            const int h = col >> 6,  d = col & 63;
            if (mode == 0) {
                obf[((size_t)(b * NHH + h) * SS + s) * HDD + d] = (__bf16)val;
            } else if (mode == 1) {
                val += 0.5f * mem[(size_t)row * HH + col];
                obf[((size_t)(b * NHH + h) * HDD + d) * SS + s] = (__bf16)val;
            } else if (mode == 2) {
                val += 0.5f * mem[(size_t)row * HH + col];
                obf[((size_t)(b * NHH + h) * SS + s) * HDD + d] = (__bf16)val;
            } else {
                val = (val * gate[0] + gbias[0]) * dyn[row];
                of32[(size_t)row * HH + col] = val;
            }
        }
    }
}

// ---------------- flash attention: block = (b, h, 64 q-rows), 4 waves -------
// K/V key tiles are shared by all 4 waves -> TDM-staged into LDS, double
// buffered (K: 64x32 from [HD][S]; V: 32x64 from [S][HD]; 4KB each).
__global__ void __launch_bounds__(128)
attn_kernel(const __bf16* __restrict__ Q, const __bf16* __restrict__ Kt,
            const __bf16* __restrict__ V, const int* __restrict__ mask,
            __bf16* __restrict__ O) {
    const int lane = threadIdx.x & 31;
    const int wave = threadIdx.x >> 5;
    const int half = lane >> 4;
    const int l16  = lane & 15;
    const int h    = blockIdx.y;
    const int b    = blockIdx.z;
    const int q0   = blockIdx.x * 64 + wave * 16;

    const __bf16* Qh  = Q  + (size_t)(b * NHH + h) * SS * HDD;
    const __bf16* Kth = Kt + (size_t)(b * NHH + h) * HDD * SS;   // [HD][S]
    const __bf16* Vh  = V  + (size_t)(b * NHH + h) * SS * HDD;   // [S][HD]
    const int* maskb  = mask + (size_t)b * SS * SS;

    __shared__ alignas(16) __bf16 ktile[2][64][32];
    __shared__ alignas(16) __bf16 vtile[2][32][64];
    __shared__ alignas(16) __bf16 pbuf[4][16][34];
    __bf16 (*pb)[34] = pbuf[wave];

    // Q A-fragments (K-chunks d=0..31 and 32..63) stay resident
    Frag qf[2];
#pragma unroll
    for (int c = 0; c < 2; ++c)
#pragma unroll
        for (int v = 0; v < 8; ++v)
            qf[c].u[v] = *reinterpret_cast<const uint32_t*>(
                Qh + (size_t)(q0 + l16) * HDD + c * 32 + a_k(v, half));

    v8f zero = {};
    v8f o[4];
#pragma unroll
    for (int i = 0; i < 4; ++i) o[i] = zero;
    float rowm[8], rowl[8];
#pragma unroll
    for (int r = 0; r < 8; ++r) { rowm[r] = -1e30f; rowl[r] = 0.0f; }

    const float SCALE = 0.125f;        // 1/sqrt(64)
    const float L2E   = 1.44269504f;

#if HAVE_TDM
    if (wave == 0) {
        tdm_load_2d(lds_off(&ktile[0][0][0]), Kth, SS, HDD, 32, 64, SS);
        tdm_load_2d(lds_off(&vtile[0][0][0]), Vh,  HDD, SS, 64, 32, HDD);
    }
#endif

    for (int kt = 0; kt < SS / 32; ++kt) {
        const int k0  = kt * 32;
        const int cur = kt & 1;
#if HAVE_TDM
        if (wave == 0) __builtin_amdgcn_s_wait_tensorcnt((short)0);
        __syncthreads();
        if (wave == 0 && k0 + 32 < SS) {
            tdm_load_2d(lds_off(&ktile[cur ^ 1][0][0]), Kth + (k0 + 32),
                        SS, HDD, 32, 64, SS);
            tdm_load_2d(lds_off(&vtile[cur ^ 1][0][0]), Vh + (size_t)(k0 + 32) * HDD,
                        HDD, SS, 64, 32, HDD);
        }
#else
        __syncthreads();
        {
            uint32_t* kdst = (uint32_t*)&ktile[cur][0][0];
            for (int i = threadIdx.x; i < 64 * 16; i += 128) {
                int r = i >> 4, c = i & 15;
                kdst[i] = *reinterpret_cast<const uint32_t*>(Kth + (size_t)r * SS + k0 + c * 2);
            }
            uint32_t* vdst = (uint32_t*)&vtile[cur][0][0];
            for (int i = threadIdx.x; i < 32 * 32; i += 128) {
                int r = i >> 5, c = i & 31;
                vdst[i] = *reinterpret_cast<const uint32_t*>(Vh + (size_t)(k0 + r) * HDD + c * 2);
            }
        }
        __syncthreads();
#endif

        // preload all 4 K fragments, then the 4 score WMMAs run back-to-back
        Frag kb[2][2];   // [t][chunk]
#pragma unroll
        for (int t = 0; t < 2; ++t)
#pragma unroll
            for (int v = 0; v < 8; ++v)
                for (int j = 0; j < 2; ++j) {
                    kb[t][0].v[2 * v + j] = ktile[cur][a_k(v, half) + j][t * 16 + l16];
                    kb[t][1].v[2 * v + j] = ktile[cur][32 + a_k(v, half) + j][t * 16 + l16];
                }
        v8f st[2];
#pragma unroll
        for (int t = 0; t < 2; ++t) {
            st[t] = wmma_bf16(qf[0], kb[t][0], zero);
            st[t] = wmma_bf16(qf[1], kb[t][1], st[t]);
        }

        // online softmax (row r lives in vgpr r of the D tiles, half-split)
#pragma unroll
        for (int r = 0; r < 8; ++r) {
            float s0 = st[0][r] * SCALE;
            float s1 = st[1][r] * SCALE;
            const int* mrow = maskb + (size_t)(q0 + r + 8 * half) * SS + k0 + l16;
            if (mrow[0]  == 0) s0 = -1e30f;
            if (mrow[16] == 0) s1 = -1e30f;
            float tmax  = rmax16(fmaxf(s0, s1));
            float mold  = rowm[r];
            float mnew  = fmaxf(mold, tmax);
            float alpha = exp2f((mold - mnew) * L2E);
            float p0    = exp2f((s0 - mnew) * L2E);
            float p1    = exp2f((s1 - mnew) * L2E);
            rowm[r] = mnew;
            rowl[r] = rowl[r] * alpha + rsum16(p0 + p1);
            o[0][r] *= alpha; o[1][r] *= alpha; o[2][r] *= alpha; o[3][r] *= alpha;
            pb[r + 8 * half][l16]      = (__bf16)p0;
            pb[r + 8 * half][16 + l16] = (__bf16)p1;
        }
        __asm__ volatile("" ::: "memory");   // LDS per-wave in-order; keep program order

        // reload P as A-fragment (C-layout -> A-layout via LDS)
        Frag pf;
#pragma unroll
        for (int v = 0; v < 8; ++v)
            pf.u[v] = *reinterpret_cast<const uint32_t*>(&pb[l16][a_k(v, half)]);

        // preload all 4 V fragments, then the 4 P*V WMMAs run back-to-back
        Frag vf[4];
#pragma unroll
        for (int nb = 0; nb < 4; ++nb)
#pragma unroll
            for (int v = 0; v < 8; ++v)
                for (int j = 0; j < 2; ++j)
                    vf[nb].v[2 * v + j] = vtile[cur][a_k(v, half) + j][nb * 16 + l16];
#pragma unroll
        for (int nb = 0; nb < 4; ++nb)
            o[nb] = wmma_bf16(pf, vf[nb], o[nb]);
    }

    // finalize: O /= l, write bf16 [B*S, H]
#pragma unroll
    for (int r = 0; r < 8; ++r) {
        const float inv = 1.0f / rowl[r];
        const size_t row = (size_t)b * SS + q0 + r + 8 * half;
#pragma unroll
        for (int nb = 0; nb < 4; ++nb)
            O[row * HH + h * HDD + nb * 16 + l16] = (__bf16)(o[nb][r] * inv);
    }
}

// ---------------------------------------------------------------------------
extern "C" void kernel_launch(void* const* d_in, const int* in_sizes, int n_in,
                              void* d_out, int out_size, void* d_ws, size_t ws_size,
                              hipStream_t stream) {
    (void)in_sizes; (void)n_in; (void)out_size; (void)ws_size;

    const float* hidden = (const float*)d_in[0];
    const float* cross  = (const float*)d_in[1];
    const int*   mask   = (const int*)d_in[2];
    const float* mem    = (const float*)d_in[3];   // [2,B,S,H]
    const float* dyn    = (const float*)d_in[4];   // [B,S,1]
    const float* Wq = (const float*)d_in[5];  const float* bq = (const float*)d_in[6];
    const float* Wk = (const float*)d_in[7];  const float* bk = (const float*)d_in[8];
    const float* Wv = (const float*)d_in[9];  const float* bv = (const float*)d_in[10];
    const float* Wo = (const float*)d_in[11]; const float* bo = (const float*)d_in[12];
    const float* gate  = (const float*)d_in[13];
    const float* gbias = (const float*)d_in[14];
    const float* lng   = (const float*)d_in[15];
    const float* lnb   = (const float*)d_in[16];

    char* p = (char*)d_ws;
    size_t o = 0;
    const size_t ACT = (size_t)MM * HH * sizeof(__bf16);   // 6.3 MB
    const size_t WMT = (size_t)HH * HH * sizeof(__bf16);   // 1.2 MB
#define WS_ALLOC(name, bytes) __bf16* name = (__bf16*)(p + o); o = (o + (bytes) + 255) & ~(size_t)255
    WS_ALLOC(xln, ACT);
    WS_ALLOC(crs, ACT);
    WS_ALLOC(wqb, WMT); WS_ALLOC(wkb, WMT); WS_ALLOC(wvb, WMT); WS_ALLOC(wob, WMT);
    WS_ALLOC(Qb,  ACT);  // [B,NH,S,HD]
    WS_ALLOC(Ktb, ACT);  // [B,NH,HD,S]
    WS_ALLOC(Vb,  ACT);  // [B,NH,S,HD]
    WS_ALLOC(AOb, ACT);  // attention out [B*S, H]
#undef WS_ALLOC

    const int nAct = MM * HH, nW = HH * HH;

    ln_kernel<<<MM, 256, 0, stream>>>(hidden, lng, lnb, xln);
    cvt_kernel<<<(nAct + 255) / 256, 256, 0, stream>>>(cross, crs, nAct);
    cvt_kernel<<<(nW + 255) / 256, 256, 0, stream>>>(Wq, wqb, nW);
    cvt_kernel<<<(nW + 255) / 256, 256, 0, stream>>>(Wk, wkb, nW);
    cvt_kernel<<<(nW + 255) / 256, 256, 0, stream>>>(Wv, wvb, nW);
    cvt_kernel<<<(nW + 255) / 256, 256, 0, stream>>>(Wo, wob, nW);

    dim3 gGemm(MM / 128, HH / 128);
    gemm_bf16_kernel<<<gGemm, 256, 0, stream>>>(xln, wqb, bq, 0, nullptr,
                                                nullptr, nullptr, nullptr, Qb, nullptr);
    gemm_bf16_kernel<<<gGemm, 256, 0, stream>>>(crs, wkb, bk, 1, mem,
                                                nullptr, nullptr, nullptr, Ktb, nullptr);
    gemm_bf16_kernel<<<gGemm, 256, 0, stream>>>(crs, wvb, bv, 2, mem + (size_t)MM * HH,
                                                nullptr, nullptr, nullptr, Vb, nullptr);

    dim3 gAttn(SS / 64, NHH, BB);
    attn_kernel<<<gAttn, 128, 0, stream>>>(Qb, Ktb, Vb, mask, AOb);

    gemm_bf16_kernel<<<gGemm, 256, 0, stream>>>(AOb, wob, bo, 3, nullptr,
                                                gate, gbias, dyn, nullptr, (float*)d_out);
}
